// GraphCDR_14370960572455
// MI455X (gfx1250) — matrix operand order repaired and checked by reading, not compiled, not run on hardware
//
#include <hip/hip_runtime.h>
#include <hip/hip_bf16.h>

#define NCC 512
#define NDD 256
#define DCI 512
#define DDI 278
#define H1D 512
#define H2D 64
#define NPAIR (NCC * NDD)

typedef __attribute__((ext_vector_type(16))) _Float16 v16h;
typedef __attribute__((ext_vector_type(8)))  _Float16 v8h;
typedef __attribute__((ext_vector_type(8)))  float    v8f;

// ---------------------------------------------------------------------------
// Kernel 1: cmat[i][n] = f16( sum_k cellpos[i][k]*W1[k][n] + b1[n] ) (512x512)
// ---------------------------------------------------------------------------
__global__ void k_cmat(const float* __restrict__ cellpos,
                       const float* __restrict__ W1,
                       const float* __restrict__ b1,
                       _Float16* __restrict__ cmat) {
    int idx = blockIdx.x * blockDim.x + threadIdx.x;   // 512*512 threads
    int i = idx >> 9, n = idx & 511;
    const float* xr = cellpos + i * DCI;
    float acc = b1[n];
#pragma unroll 4
    for (int k = 0; k < DCI; ++k)
        acc = fmaf(xr[k], W1[k * 512 + n], acc);
    cmat[idx] = (_Float16)acc;
}

// ---------------------------------------------------------------------------
// Kernel 2: dmat[j][n] = f16( sum_k drugpos[j][k]*W1[512+k][n] )    (256x512)
// ---------------------------------------------------------------------------
__global__ void k_dmat(const float* __restrict__ drugpos,
                       const float* __restrict__ W1,
                       _Float16* __restrict__ dmat) {
    int idx = blockIdx.x * blockDim.x + threadIdx.x;   // 256*512 threads
    int j = idx >> 9, n = idx & 511;
    const float* xr = drugpos + j * DDI;
    float acc = 0.f;
#pragma unroll 2
    for (int k = 0; k < DDI; ++k)
        acc = fmaf(xr[k], W1[(DCI + k) * 512 + n], acc);
    dmat[idx] = (_Float16)acc;
}

// ---------------------------------------------------------------------------
// Kernel 3: pack W2 [512,64] f32 -> f16 WMMA B-fragments.
// Fragment (kb, nb): 32x16 (KxN). Per-lane ISA layout (16-bit B):
//   lane L holds col n = L&15; halves 0..7 -> K = (L>>4)*8 .. +7, halves 8..15 -> K = 16 + (L>>4)*8 .. +7.
// Storage: bp[(((kb*4+nb)*32 + lane)*16 + i)]
// ---------------------------------------------------------------------------
__global__ void k_packW2(const float* __restrict__ W2, _Float16* __restrict__ bp) {
    int idx = blockIdx.x * blockDim.x + threadIdx.x;   // 512*64 threads
    int k = idx >> 6, n = idx & 63;
    int kb = k >> 5, kin = k & 31;
    int half = (kin >> 3) & 1, i = (kin & 7) + ((kin >> 4) << 3);
    int nb = n >> 4, nn = n & 15;
    int lane = nn + half * 16;
    int dst = ((kb * 4 + nb) * 32 + lane) * 16 + i;
    bp[dst] = (_Float16)W2[idx];
}

// ---------------------------------------------------------------------------
// Build one A-fragment (16x32 f16, ISA layout) with packed f16 math:
//   halves 0..7  -> K = ka  .. ka+7   (relu(c+d), v_pk_add/pk_max)
//   halves 8..15 -> K = kbb .. kbb+7
// ---------------------------------------------------------------------------
__device__ __forceinline__ v16h build_a(const _Float16* __restrict__ crow,
                                        const _Float16* __restrict__ drow,
                                        int ka, int kbb) {
    v8h da = *(const v8h*)(drow + ka);
    v8h db = *(const v8h*)(drow + kbb);
    v8h ca = *(const v8h*)(crow + ka);
    v8h cb = *(const v8h*)(crow + kbb);
    v8h z  = {};
    v8h ha = __builtin_elementwise_max(ca + da, z);
    v8h hb = __builtin_elementwise_max(cb + db, z);
    return __builtin_shufflevector(ha, hb,
                                   0, 1, 2, 3, 4, 5, 6, 7,
                                   8, 9, 10, 11, 12, 13, 14, 15);
}

// ---------------------------------------------------------------------------
// Main fused kernel: per wave, 32 pairs (two 16-row A tiles) x 64 outputs.
// Both tiles share the same 4 LDS B-fragments per K-step -> 8 WMMAs per
// 8 ds_loads, two independent accumulation chains for latency hiding.
// ---------------------------------------------------------------------------
__global__ void __launch_bounds__(256)
k_main(const _Float16* __restrict__ cmat,
       const _Float16* __restrict__ dmat,
       const _Float16* __restrict__ bpack,
       const float* __restrict__ b2,
       const float* __restrict__ W3,
       const float* __restrict__ b3,
       float* __restrict__ out_pos,
       float* __restrict__ out_feats) {
    extern __shared__ _Float16 ldsB[];   // 32768 halves = 64 KB

    // ---- async stage of all 64 B-fragments into LDS ----
    {
        const unsigned lbase = (unsigned)(uintptr_t)&ldsB[0];
        const unsigned long long gb = (unsigned long long)(uintptr_t)bpack;
#pragma unroll
        for (int it = 0; it < 16; ++it) {
            unsigned idx  = threadIdx.x + it * 256u;   // 4096 x 16B = 64KB
            unsigned laddr = lbase + idx * 16u;
            unsigned voff  = idx * 16u;
            asm volatile("global_load_async_to_lds_b128 %0, %1, %2"
                         :: "v"(laddr), "v"(voff), "s"(gb)
                         : "memory");
        }
        asm volatile("s_wait_asynccnt 0x0" ::: "memory");
    }
    __syncthreads();

    const int lane  = threadIdx.x & 31;
    const int wave  = threadIdx.x >> 5;
    const int pair0 = (blockIdx.x * 8 + wave) * 32;   // 32 consecutive pairs
    const int row   = lane & 15;
    const int khalf = lane >> 4;
    // 32 consecutive pairs share one cell (256 drugs per cell, 256%32==0)
    const _Float16* crow  = cmat + (size_t)(pair0 >> 8) * H1D;
    const _Float16* drowA = dmat + (size_t)((pair0 & 255) + row) * H1D;
    const _Float16* drowB = drowA + 16 * H1D;

    v8f accA0 = {}, accA1 = {}, accA2 = {}, accA3 = {};
    v8f accB0 = {}, accB1 = {}, accB2 = {}, accB3 = {};
    const v16h* bfrag = (const v16h*)ldsB;

    for (int kb = 0; kb < 16; ++kb) {
        const int ka  = kb * 32 + khalf * 8;
        const int kbb = ka + 16;

        v16h a0 = build_a(crow, drowA, ka, kbb);
        v16h a1 = build_a(crow, drowB, ka, kbb);

        v16h bv0 = bfrag[(kb * 4 + 0) * 32 + lane];
        v16h bv1 = bfrag[(kb * 4 + 1) * 32 + lane];
        v16h bv2 = bfrag[(kb * 4 + 2) * 32 + lane];
        v16h bv3 = bfrag[(kb * 4 + 3) * 32 + lane];

        accA0 = __builtin_amdgcn_wmma_f32_16x16x32_f16(false, a0, false, bv0, (short)0, accA0, false, false);
        accB0 = __builtin_amdgcn_wmma_f32_16x16x32_f16(false, a1, false, bv0, (short)0, accB0, false, false);
        accA1 = __builtin_amdgcn_wmma_f32_16x16x32_f16(false, a0, false, bv1, (short)0, accA1, false, false);
        accB1 = __builtin_amdgcn_wmma_f32_16x16x32_f16(false, a1, false, bv1, (short)0, accB1, false, false);
        accA2 = __builtin_amdgcn_wmma_f32_16x16x32_f16(false, a0, false, bv2, (short)0, accA2, false, false);
        accB2 = __builtin_amdgcn_wmma_f32_16x16x32_f16(false, a1, false, bv2, (short)0, accB2, false, false);
        accA3 = __builtin_amdgcn_wmma_f32_16x16x32_f16(false, a0, false, bv3, (short)0, accA3, false, false);
        accB3 = __builtin_amdgcn_wmma_f32_16x16x32_f16(false, a1, false, bv3, (short)0, accB3, false, false);
    }

    // ---- epilogue: bias + relu, store feats, W3 dot, sigmoid ----
    const int n = lane & 15;
    const float b2v0 = b2[n],      b2v1 = b2[16 + n];
    const float b2v2 = b2[32 + n], b2v3 = b2[48 + n];
    const float w30 = W3[n],       w31 = W3[16 + n];
    const float w32 = W3[32 + n],  w33 = W3[48 + n];
    const float b3v = b3[0];

    float sjA[8], sjB[8];
#pragma unroll
    for (int j = 0; j < 8; ++j) {
        int m = khalf * 8 + j;          // C/D layout: M = (lane>>4)*8 + j, N = lane&15
        float* frA = out_feats + (size_t)(pair0 + m) * 64;
        float* frB = out_feats + (size_t)(pair0 + 16 + m) * 64;
        float vA0 = fmaxf(accA0[j] + b2v0, 0.f);
        float vA1 = fmaxf(accA1[j] + b2v1, 0.f);
        float vA2 = fmaxf(accA2[j] + b2v2, 0.f);
        float vA3 = fmaxf(accA3[j] + b2v3, 0.f);
        float vB0 = fmaxf(accB0[j] + b2v0, 0.f);
        float vB1 = fmaxf(accB1[j] + b2v1, 0.f);
        float vB2 = fmaxf(accB2[j] + b2v2, 0.f);
        float vB3 = fmaxf(accB3[j] + b2v3, 0.f);
        frA[n] = vA0; frA[16 + n] = vA1; frA[32 + n] = vA2; frA[48 + n] = vA3;
        frB[n] = vB0; frB[16 + n] = vB1; frB[32 + n] = vB2; frB[48 + n] = vB3;
        sjA[j] = fmaf(vA0, w30, fmaf(vA1, w31, fmaf(vA2, w32, vA3 * w33)));
        sjB[j] = fmaf(vB0, w30, fmaf(vB1, w31, fmaf(vB2, w32, vB3 * w33)));
    }
#pragma unroll
    for (int j = 0; j < 8; ++j) {
        float sA = sjA[j];
        float sB = sjB[j];
        // reduce over the 16 lanes sharing this row (xor masks stay in half)
        sA += __shfl_xor(sA, 1, 32);  sB += __shfl_xor(sB, 1, 32);
        sA += __shfl_xor(sA, 2, 32);  sB += __shfl_xor(sB, 2, 32);
        sA += __shfl_xor(sA, 4, 32);  sB += __shfl_xor(sB, 4, 32);
        sA += __shfl_xor(sA, 8, 32);  sB += __shfl_xor(sB, 8, 32);
        if ((lane & 15) == 0) {
            int m = khalf * 8 + j;
            out_pos[pair0 + m]      = 1.f / (1.f + __expf(-(sA + b3v)));
            out_pos[pair0 + 16 + m] = 1.f / (1.f + __expf(-(sB + b3v)));
        }
    }
}

// ---------------------------------------------------------------------------
extern "C" void kernel_launch(void* const* d_in, const int* in_sizes, int n_in,
                              void* d_out, int out_size, void* d_ws, size_t ws_size,
                              hipStream_t stream) {
    const float* cellpos = (const float*)d_in[0];
    const float* drugpos = (const float*)d_in[1];
    const float* W1      = (const float*)d_in[2];
    const float* b1      = (const float*)d_in[3];
    const float* W2      = (const float*)d_in[4];
    const float* b2      = (const float*)d_in[5];
    const float* W3      = (const float*)d_in[6];
    const float* b3      = (const float*)d_in[7];

    // workspace layout (f16): cmat 512KB | dmat 256KB | packed W2 64KB
    _Float16* cmat  = (_Float16*)d_ws;
    _Float16* dmat  = cmat + NCC * H1D;
    _Float16* bpack = dmat + NDD * H1D;

    float* out_pos   = (float*)d_out;
    float* out_feats = out_pos + NPAIR;

    k_cmat  <<<(NCC * H1D) / 256, 256, 0, stream>>>(cellpos, W1, b1, cmat);
    k_dmat  <<<(NDD * H1D) / 256, 256, 0, stream>>>(drugpos, W1, dmat);
    k_packW2<<<(H1D * H2D) / 256, 256, 0, stream>>>(W2, bpack);

    // 131072 pairs / (8 waves * 32 rows) = 512 blocks; 64KB dynamic LDS for B
    k_main<<<NPAIR / 256, 256, 65536, stream>>>(cmat, dmat, bpack,
                                                b2, W3, b3, out_pos, out_feats);
}
